// GLALayer_37314675868063
// MI455X (gfx1250) — matrix operand order repaired and checked by reading, not compiled, not run on hardware
//
#include <hip/hip_runtime.h>
#include <hip/hip_bf16.h>

// GLA layer for MI455X (gfx1250). All GEMMs via v_wmma_f32_16x16x32_bf16,
// double-buffered LDS staging with global_load_async_to_lds_b128,
// back-to-back WMMA issue (all B fragments resident in VGPRs).
// B=4, T=4096, DIM=2048, H=16, HD=128.

#define B_   4
#define T_   4096
#define DIM_ 2048
#define H_   16
#define HD_  128
#define BT_  (B_ * T_)

typedef __attribute__((ext_vector_type(16))) __bf16 v16bf;
typedef __attribute__((ext_vector_type(8)))  float  v8f;

typedef int v4i __attribute__((vector_size(16)));
typedef __attribute__((address_space(1))) v4i as1_v4i;
typedef __attribute__((address_space(3))) v4i as3_v4i;

#if defined(__gfx1250__) && __has_builtin(__builtin_amdgcn_global_load_async_to_lds_b128) && __has_builtin(__builtin_amdgcn_s_wait_asynccnt)
#define USE_ASYNC_LDS 1
#else
#define USE_ASYNC_LDS 0
#endif

// ---------- bf16 helpers (bit-level, round-to-nearest-even) ----------
__device__ __forceinline__ unsigned short f2bf(float f) {
    unsigned u = __float_as_uint(f);
    unsigned r = u + 0x7FFFu + ((u >> 16) & 1u);
    return (unsigned short)(r >> 16);
}
__device__ __forceinline__ float bf2f(unsigned short s) {
    return __uint_as_float(((unsigned)s) << 16);
}

union FragU { v16bf v; uint4 q[2]; };

// A fragment (16x32) from a ROW-MAJOR LDS tile [row][k], stride ld elems.
__device__ __forceinline__ v16bf frag_a(const unsigned short* tile, int row_base, int ld, int lane) {
    const int m = lane & 15, half = lane >> 4;
    const unsigned short* p = tile + (row_base + m) * ld + half * 8;
    FragU f;
    f.q[0] = *(const uint4*)(p);
    f.q[1] = *(const uint4*)(p + 16);
    return f.v;
}

// B fragment (32x16) from a TRANSPOSED LDS tile Bt[n][kk], stride ld elems.
__device__ __forceinline__ v16bf frag_b(const unsigned short* tileT, int ld, int nb, int k_off, int lane) {
    const int n = lane & 15, half = lane >> 4;
    const unsigned short* p = tileT + (nb + n) * ld + k_off + half * 8;
    FragU f;
    f.q[0] = *(const uint4*)(p);
    f.q[1] = *(const uint4*)(p + 16);
    return f.v;
}

// Transpose-scatter one uint4 (8 bf16 along columns) into Bt[col][row].
__device__ __forceinline__ void scatter8(unsigned short* tileT, int ld, int colbase, int row, uint4 g) {
    union { uint4 q; unsigned short s[8]; } t; t.q = g;
#pragma unroll
    for (int s2 = 0; s2 < 8; ++s2) tileT[(colbase + s2) * ld + row] = t.s[s2];
}

// Straight 16B copy global->LDS on the gfx1250 async engine.
__device__ __forceinline__ void copy16(const unsigned short* gsrc, unsigned short* ldst) {
#if USE_ASYNC_LDS
    __builtin_amdgcn_global_load_async_to_lds_b128(
        (as1_v4i*)(void*)gsrc, (as3_v4i*)(void*)ldst, 0, 0);
#else
    *(uint4*)ldst = *(const uint4*)gsrc;
#endif
}
__device__ __forceinline__ void copy_fence() {
#if USE_ASYNC_LDS
    __builtin_amdgcn_s_wait_asynccnt(0);
#endif
}

// 8 WMMAs over a full 128-wide N strip. All 8 B fragments are loaded first
// (16x ds_load_b128 into distinct VGPRs), then the WMMAs issue back-to-back.
__device__ __forceinline__ void wmma_strip(v8f (&acc)[8], v16bf af,
                                           const unsigned short* tileT, int ld, int k_off, int lane) {
    v16bf bf[8];
#pragma unroll
    for (int nt = 0; nt < 8; ++nt) bf[nt] = frag_b(tileT, ld, nt * 16, k_off, lane);
#pragma unroll
    for (int nt = 0; nt < 8; ++nt)
        acc[nt] = __builtin_amdgcn_wmma_f32_16x16x32_bf16(
            false, af, false, bf[nt], (short)0, acc[nt], false, false);
}

// ---------- fp32 -> bf16 cast ----------
__global__ __launch_bounds__(256) void cast_f32_bf16(const float* __restrict__ in,
                                                     unsigned short* __restrict__ out, long n) {
    long i = ((long)blockIdx.x * 256 + threadIdx.x) * 4;
    if (i + 3 < n) {
        float4 f = *(const float4*)&in[i];
        unsigned lo = (unsigned)f2bf(f.x) | ((unsigned)f2bf(f.y) << 16);
        unsigned hi = (unsigned)f2bf(f.z) | ((unsigned)f2bf(f.w) << 16);
        *(uint2*)&out[i] = make_uint2(lo, hi);
    }
}

// ---------- bf16 WMMA GEMM: C = A*B + bias; BM=128,BN=128,BK=32, double-buffered ----------
template<int OUT_BF16>
__global__ __launch_bounds__(256) void gemm_bf16_kernel(
    const unsigned short* __restrict__ A, const unsigned short* __restrict__ Bm,
    const float* __restrict__ bias, void* __restrict__ Cout,
    int M, int N, int K)
{
    __shared__ unsigned short As[2][128 * 40];
    __shared__ unsigned short Bt[2][128 * 40];
    const int tid  = threadIdx.x;
    const int lane = tid & 31;
    const int wv   = tid >> 5;
    const int half = lane >> 4;
    const int m0   = blockIdx.y * 128;
    const int n0   = blockIdx.x * 128;

    // prologue: stage tile 0 into buffer 0
#pragma unroll
    for (int i = 0; i < 2; ++i) {
        int v = tid + 256 * i;
        int row = v >> 2, j = v & 3;
        copy16(&A[(size_t)(m0 + row) * K + j * 8], &As[0][row * 40 + j * 8]);
    }
#pragma unroll
    for (int i = 0; i < 2; ++i) {
        int v = tid + 256 * i;
        int kk = v >> 4, j = v & 15;
        uint4 g = *(const uint4*)&Bm[(size_t)kk * N + n0 + j * 8];
        scatter8(&Bt[0][0], 40, j * 8, kk, g);
    }
    copy_fence();
    __syncthreads();

    v8f acc[8] = {};
    for (int k0 = 0; k0 < K; k0 += 32) {
        const int cur = (k0 >> 5) & 1, nxt = cur ^ 1;
        const bool more = (k0 + 32 < K);
        uint4 gB[2];
        if (more) {
#pragma unroll
            for (int i = 0; i < 2; ++i) {        // async A for tile k+1
                int v = tid + 256 * i;
                int row = v >> 2, j = v & 3;
                copy16(&A[(size_t)(m0 + row) * K + k0 + 32 + j * 8],
                       &As[nxt][row * 40 + j * 8]);
            }
#pragma unroll
            for (int i = 0; i < 2; ++i) {        // B tile k+1 -> registers
                int v = tid + 256 * i;
                int kk = v >> 4, j = v & 15;
                gB[i] = *(const uint4*)&Bm[(size_t)(k0 + 32 + kk) * N + n0 + j * 8];
            }
            if (k0 + 64 < K)
                __builtin_prefetch(&A[(size_t)(m0 + (tid >> 1)) * K + k0 + 64], 0, 0);
        }
        // compute on current tile while next tile is in flight
        v16bf af = frag_a(&As[cur][0], 16 * wv, 40, lane);
        wmma_strip(acc, af, &Bt[cur][0], 40, 0, lane);
        if (more) {
#pragma unroll
            for (int i = 0; i < 2; ++i) {
                int v = tid + 256 * i;
                int kk = v >> 4, j = v & 15;
                scatter8(&Bt[nxt][0], 40, j * 8, kk, gB[i]);
            }
            copy_fence();
        }
        __syncthreads();
    }
#pragma unroll
    for (int nt = 0; nt < 8; ++nt) {
#pragma unroll
        for (int r = 0; r < 8; ++r) {
            int row = m0 + 16 * wv + r + half * 8;
            int col = n0 + nt * 16 + (lane & 15);
            float val = acc[nt][r] + bias[col];
            if (OUT_BF16)
                ((unsigned short*)Cout)[(size_t)row * N + col] = f2bf(val);
            else
                ((float*)Cout)[(size_t)row * N + col] = val;
        }
    }
}

// ---------- double softmax over HD=128 on K (in-place, bf16), one wave per row ----------
__global__ __launch_bounds__(256) void ksoftmax_kernel(unsigned short* __restrict__ qkv,
                                                       const int* __restrict__ mask) {
    const int lane = threadIdx.x & 31;
    const int gid  = blockIdx.x * 8 + (threadIdx.x >> 5);
    const int h  = gid & 15;
    const int bt = gid >> 4;
    size_t base = (size_t)bt * (3 * DIM_) + DIM_ + (size_t)h * HD_;

    float x[4];
#pragma unroll
    for (int i = 0; i < 4; ++i) x[i] = bf2f(qkv[base + lane + 32 * i]);

#pragma unroll
    for (int pass = 0; pass < 2; ++pass) {
        float mx = fmaxf(fmaxf(x[0], x[1]), fmaxf(x[2], x[3]));
#pragma unroll
        for (int off = 16; off > 0; off >>= 1) mx = fmaxf(mx, __shfl_xor(mx, off, 32));
        float s = 0.f;
#pragma unroll
        for (int i = 0; i < 4; ++i) { x[i] = __expf(x[i] - mx); s += x[i]; }
#pragma unroll
        for (int off = 16; off > 0; off >>= 1) s += __shfl_xor(s, off, 32);
        float inv = 1.0f / s;
#pragma unroll
        for (int i = 0; i < 4; ++i) x[i] *= inv;
        if (pass == 0) {
            float pen = (1.0f - (float)mask[bt]) * -1e9f;
#pragma unroll
            for (int i = 0; i < 4; ++i) x[i] += pen;
        }
    }
#pragma unroll
    for (int i = 0; i < 4; ++i) qkv[base + lane + 32 * i] = f2bf(x[i]);
}

// ---------- context = K^T @ V per (b,h), double-buffered transposed tiles ----------
__global__ __launch_bounds__(256) void ctx_kernel(const unsigned short* __restrict__ qkv,
                                                  unsigned short* __restrict__ ctx) {
    const int bh = blockIdx.x;
    const int b = bh >> 4, h = bh & 15;
    __shared__ unsigned short Kt[2][128 * 40];   // [d][t]
    __shared__ unsigned short Vt[2][128 * 40];   // [e][t]
    const int tid = threadIdx.x, lane = tid & 31, wv = tid >> 5, half = lane >> 4;

    // prologue: stage t0 = 0 into buffer 0
#pragma unroll
    for (int i = 0; i < 2; ++i) {
        int v = tid + 256 * i;
        int trow = v >> 4, j = v & 15;
        size_t base = (size_t)(b * T_ + trow) * (3 * DIM_) + DIM_ + h * HD_ + j * 8;
        scatter8(&Kt[0][0], 40, j * 8, trow, *(const uint4*)&qkv[base]);
        scatter8(&Vt[0][0], 40, j * 8, trow, *(const uint4*)&qkv[base + DIM_]);
    }
    __syncthreads();

    v8f acc[8] = {};
    for (int t0 = 0; t0 < T_; t0 += 32) {
        const int cur = (t0 >> 5) & 1, nxt = cur ^ 1;
        const bool more = (t0 + 32 < T_);
        uint4 gK[2], gV[2];
        if (more) {
#pragma unroll
            for (int i = 0; i < 2; ++i) {
                int v = tid + 256 * i;
                int trow = v >> 4, j = v & 15;
                size_t base = (size_t)(b * T_ + t0 + 32 + trow) * (3 * DIM_) + DIM_ + h * HD_ + j * 8;
                gK[i] = *(const uint4*)&qkv[base];
                gV[i] = *(const uint4*)&qkv[base + DIM_];
            }
            if (t0 + 64 < T_)
                __builtin_prefetch(&qkv[(size_t)(b * T_ + t0 + 64 + (tid >> 4)) * (3 * DIM_) + DIM_ + h * HD_], 0, 0);
        }
        v16bf af = frag_a(&Kt[cur][0], 16 * wv, 40, lane);
        wmma_strip(acc, af, &Vt[cur][0], 40, 0, lane);
        if (more) {
#pragma unroll
            for (int i = 0; i < 2; ++i) {
                int v = tid + 256 * i;
                int trow = v >> 4, j = v & 15;
                scatter8(&Kt[nxt][0], 40, j * 8, trow, gK[i]);
                scatter8(&Vt[nxt][0], 40, j * 8, trow, gV[i]);
            }
        }
        __syncthreads();
    }
#pragma unroll
    for (int nt = 0; nt < 8; ++nt)
#pragma unroll
        for (int r = 0; r < 8; ++r) {
            int d = 16 * wv + r + half * 8;
            int e = nt * 16 + (lane & 15);
            ctx[((size_t)bh * HD_ + d) * HD_ + e] = f2bf(acc[nt][r]);
        }
}

// ---------- out = Q @ ctx per (b,h): [128 t-rows, 128] per block ----------
__global__ __launch_bounds__(256) void qctx_kernel(const unsigned short* __restrict__ qkv,
                                                   const unsigned short* __restrict__ ctx,
                                                   unsigned short* __restrict__ attn) {
    const int bh = blockIdx.x;
    const int b = bh >> 4, h = bh & 15;
    const int t0 = blockIdx.y * 128;
    __shared__ unsigned short Ct[128 * 136];     // transposed ctx, resident
    __shared__ unsigned short Qs[2][128 * 40];   // ping-pong Q tiles
    const int tid = threadIdx.x, lane = tid & 31, wv = tid >> 5, half = lane >> 4;

#pragma unroll
    for (int i = 0; i < 8; ++i) {                // stage full ctx, transposed, once
        int v = tid + 256 * i;
        int drow = v >> 4, j = v & 15;
        uint4 g = *(const uint4*)&ctx[(size_t)bh * (HD_ * HD_) + drow * HD_ + j * 8];
        scatter8(&Ct[0], 136, j * 8, drow, g);
    }
#pragma unroll
    for (int i = 0; i < 2; ++i) {                // stage Q tile k0=0 (async)
        int v = tid + 256 * i;
        int row = v >> 2, j = v & 3;
        copy16(&qkv[(size_t)(b * T_ + t0 + row) * (3 * DIM_) + h * HD_ + j * 8],
               &Qs[0][row * 40 + j * 8]);
    }
    copy_fence();
    __syncthreads();

    v8f acc[8] = {};
    for (int k0 = 0; k0 < HD_; k0 += 32) {
        const int cur = (k0 >> 5) & 1, nxt = cur ^ 1;
        const bool more = (k0 + 32 < HD_);
        if (more) {
#pragma unroll
            for (int i = 0; i < 2; ++i) {        // async Q tile k+1
                int v = tid + 256 * i;
                int row = v >> 2, j = v & 3;
                copy16(&qkv[(size_t)(b * T_ + t0 + row) * (3 * DIM_) + h * HD_ + k0 + 32 + j * 8],
                       &Qs[nxt][row * 40 + j * 8]);
            }
        }
        v16bf af = frag_a(&Qs[cur][0], 16 * wv, 40, lane);
        wmma_strip(acc, af, &Ct[0], 136, k0, lane);
        if (more) copy_fence();
        __syncthreads();
    }
#pragma unroll
    for (int nt = 0; nt < 8; ++nt)
#pragma unroll
        for (int r = 0; r < 8; ++r) {
            int t = t0 + 16 * wv + r + half * 8;
            int e = nt * 16 + (lane & 15);
            attn[(size_t)(b * T_ + t) * DIM_ + h * HD_ + e] = f2bf(acc[nt][r]);
        }
}

extern "C" void kernel_launch(void* const* d_in, const int* in_sizes, int n_in,
                              void* d_out, int out_size, void* d_ws, size_t ws_size,
                              hipStream_t stream) {
    const float* X    = (const float*)d_in[0];
    const int*   mask = (const int*)d_in[1];
    const float* Wqkv = (const float*)d_in[2];
    const float* bqkv = (const float*)d_in[3];
    const float* Wout = (const float*)d_in[4];
    const float* bout = (const float*)d_in[5];

    char* ws = (char*)d_ws;
    unsigned short* Xbf     = (unsigned short*)ws; ws += (size_t)BT_ * DIM_ * 2;
    unsigned short* Wqkv_bf = (unsigned short*)ws; ws += (size_t)DIM_ * 3 * DIM_ * 2;
    unsigned short* Wout_bf = (unsigned short*)ws; ws += (size_t)DIM_ * DIM_ * 2;
    unsigned short* qkv_bf  = (unsigned short*)ws; ws += (size_t)BT_ * 3 * DIM_ * 2;
    unsigned short* ctx_bf  = (unsigned short*)ws; ws += (size_t)B_ * H_ * HD_ * HD_ * 2;
    unsigned short* attn_bf = (unsigned short*)ws;

    const long nX  = (long)BT_ * DIM_;
    const long nWq = (long)DIM_ * 3 * DIM_;
    const long nWo = (long)DIM_ * DIM_;
    cast_f32_bf16<<<(unsigned)((nX  / 4 + 255) / 256), 256, 0, stream>>>(X,    Xbf,     nX);
    cast_f32_bf16<<<(unsigned)((nWq / 4 + 255) / 256), 256, 0, stream>>>(Wqkv, Wqkv_bf, nWq);
    cast_f32_bf16<<<(unsigned)((nWo / 4 + 255) / 256), 256, 0, stream>>>(Wout, Wout_bf, nWo);

    gemm_bf16_kernel<1><<<dim3(3 * DIM_ / 128, BT_ / 128), 256, 0, stream>>>(
        Xbf, Wqkv_bf, bqkv, qkv_bf, BT_, 3 * DIM_, DIM_);

    ksoftmax_kernel<<<BT_ * H_ / 8, 256, 0, stream>>>(qkv_bf, mask);

    ctx_kernel<<<B_ * H_, 256, 0, stream>>>(qkv_bf, ctx_bf);

    qctx_kernel<<<dim3(B_ * H_, T_ / 128), 256, 0, stream>>>(qkv_bf, ctx_bf, attn_bf);

    gemm_bf16_kernel<0><<<dim3(DIM_ / 128, BT_ / 128), 256, 0, stream>>>(
        attn_bf, Wout_bf, bout, d_out, BT_, DIM_, DIM_);
}
